// MultiInterestExtractor_40123584479487
// MI455X (gfx1250) — compile-verified, matrix operand order checked by reading, not run on hardware
//
#include <hip/hip_runtime.h>
#include <hip/hip_bf16.h>
#include <math.h>

typedef __attribute__((ext_vector_type(16))) _Float16 v16h;
typedef __attribute__((ext_vector_type(8)))  _Float16 v8h;
typedef __attribute__((ext_vector_type(8)))  float    v8f;

#define LL 200
#define DD 64
#define AA 5
#define KK 4
#define LP 224      // L padded to 7 chunks of 32 for WMMA K-dim
#define STR 232     // f16 row stride: multiple of 8 -> every 8-elem col offset is 16B aligned

__global__ __launch_bounds__(32)
void mie_kernel(const float* __restrict__ behavior,
                const int*   __restrict__ seqlen,
                const float* __restrict__ W1w,
                const float* __restrict__ W1b,
                const float* __restrict__ W2w,
                const float* __restrict__ W2b,
                float*       __restrict__ out)
{
    __shared__ __align__(16) _Float16 hT[DD * STR];   // h transposed: [d][l], f16
    __shared__ __align__(16) _Float16 pL[KK * STR];   // attention probs: [k][l], f16
    __shared__ float w1[AA * DD];
    __shared__ float w1b[AA];
    __shared__ float w2[KK * AA];
    __shared__ float w2b[KK];

    const int n    = blockIdx.x;
    const int lane = threadIdx.x;   // wave32: lane == threadIdx.x

    // ---------------- stage tiny weights into LDS ----------------
    for (int i = lane; i < AA * DD; i += 32) w1[i] = W1w[i];
    if (lane < AA)      w1b[lane] = W1b[lane];
    if (lane < KK * AA) w2[lane]  = W2w[lane];
    if (lane < KK)      w2b[lane] = W2b[lane];

    // ---------------- stream h[n] (200x64 f32) -> LDS f16, transposed --------
    const float4* hp = (const float4*)(behavior + (size_t)n * (LL * DD));
    for (int i = lane; i < (LL * DD) / 4; i += 32) {
        float4 v = hp[i];
        int l  = i >> 4;            // 16 float4 per 64-wide row
        int d0 = (i & 15) << 2;
        hT[(d0 + 0) * STR + l] = (_Float16)v.x;
        hT[(d0 + 1) * STR + l] = (_Float16)v.y;
        hT[(d0 + 2) * STR + l] = (_Float16)v.z;
        hT[(d0 + 3) * STR + l] = (_Float16)v.w;
    }
    // zero l in [200,224) so WMMA B-operand pad reads are clean (never NaN)
    {
        v8h z = {0, 0, 0, 0, 0, 0, 0, 0};
        for (int d = lane; d < DD; d += 32) {
            *(v8h*)&hT[d * STR + 200] = z;
            *(v8h*)&hT[d * STR + 208] = z;
            *(v8h*)&hT[d * STR + 216] = z;
        }
    }
    __syncthreads();

    // ---------------- scores: each lane owns l = lane + 32*i, i<7 ------------
    const int slen = seqlen[n];
    float accm[7][AA];
    #pragma unroll
    for (int i = 0; i < 7; ++i)
        #pragma unroll
        for (int a = 0; a < AA; ++a) accm[i][a] = w1b[a];

    for (int d = 0; d < DD; ++d) {
        float wr[AA];
        #pragma unroll
        for (int a = 0; a < AA; ++a) wr[a] = w1[a * DD + d];
        #pragma unroll
        for (int i = 0; i < 7; ++i) {
            float hv = (float)hT[d * STR + lane + 32 * i];   // pad cols read 0
            #pragma unroll
            for (int a = 0; a < AA; ++a)
                accm[i][a] = fmaf(hv, wr[a], accm[i][a]);
        }
    }

    float sv[KK][7];
    #pragma unroll
    for (int i = 0; i < 7; ++i) {
        const int l = lane + 32 * i;
        float t[AA];
        #pragma unroll
        for (int a = 0; a < AA; ++a) t[a] = tanhf(accm[i][a]);
        const bool valid = (l < slen);                        // l>=200 auto-invalid
        #pragma unroll
        for (int k = 0; k < KK; ++k) {
            float s = w2b[k];
            #pragma unroll
            for (int a = 0; a < AA; ++a) s = fmaf(t[a], w2[k * AA + a], s);
            sv[k][i] = valid ? s : -INFINITY;
        }
    }

    // ---------------- per-row softmax over l (wave32 reductions) -------------
    #pragma unroll
    for (int k = 0; k < KK; ++k) {
        float mx = -INFINITY;
        #pragma unroll
        for (int i = 0; i < 7; ++i) mx = fmaxf(mx, sv[k][i]);
        for (int off = 16; off > 0; off >>= 1)
            mx = fmaxf(mx, __shfl_xor(mx, off, 32));
        float ev[7];
        float sum = 0.f;
        #pragma unroll
        for (int i = 0; i < 7; ++i) { ev[i] = expf(sv[k][i] - mx); sum += ev[i]; }
        for (int off = 16; off > 0; off >>= 1)
            sum += __shfl_xor(sum, off, 32);
        const float inv = 1.0f / sum;                         // slen>=1 -> sum>0
        #pragma unroll
        for (int i = 0; i < 7; ++i)
            pL[k * STR + lane + 32 * i] = (_Float16)(ev[i] * inv);
    }
    __syncthreads();

    // ---------------- out[k,d] = sum_l p[k,l]*h[l,d] via v_wmma --------------
    // M=k (rows 0..3 live), N=d tile of 16, K-dim = l chunk of 32.
    // Lane layout (16-bit operands): lane t holds row t%16; K slots
    // {0..7,16..23} for lanes<16, {8..15,24..31} for lanes>=16 -> two
    // contiguous 8-element (16B) runs per chunk => ds_load_b128 pairs.
    const int row = lane & 15;
    const int g   = lane >> 4;
    v8f zf = {0, 0, 0, 0, 0, 0, 0, 0};
    v8f acc[4] = {zf, zf, zf, zf};
    v16h zero16 = {0,0,0,0,0,0,0,0,0,0,0,0,0,0,0,0};

    #pragma unroll
    for (int c = 0; c < 7; ++c) {
        const int l0 = c * 32 + g * 8;
        v16h av;
        if (row < KK) {
            v8h alo = *(const v8h*)&pL[row * STR + l0];
            v8h ahi = *(const v8h*)&pL[row * STR + l0 + 16];
            av = __builtin_shufflevector(alo, ahi,
                 0,1,2,3,4,5,6,7,8,9,10,11,12,13,14,15);
        } else {
            av = zero16;                                      // dead A rows
        }
        #pragma unroll
        for (int dt = 0; dt < 4; ++dt) {
            const int d = dt * 16 + row;
            v8h blo = *(const v8h*)&hT[d * STR + l0];
            v8h bhi = *(const v8h*)&hT[d * STR + l0 + 16];
            v16h bv = __builtin_shufflevector(blo, bhi,
                      0,1,2,3,4,5,6,7,8,9,10,11,12,13,14,15);
            acc[dt] = __builtin_amdgcn_wmma_f32_16x16x32_f16(
                          false, av, false, bv, (short)0, acc[dt], false, false);
        }
    }

    // D layout: lane t holds C[M = 8*(t/16)+j][N = t%16]; valid k=0..3 live in
    // lanes 0..15, VGPR slots j=0..3.
    if (g == 0) {
        float* op = out + (size_t)n * (KK * DD);
        #pragma unroll
        for (int dt = 0; dt < 4; ++dt)
            #pragma unroll
            for (int j = 0; j < KK; ++j)
                op[j * DD + dt * 16 + row] = acc[dt][j];
    }
}

extern "C" void kernel_launch(void* const* d_in, const int* in_sizes, int n_in,
                              void* d_out, int out_size, void* d_ws, size_t ws_size,
                              hipStream_t stream) {
    const float* behavior = (const float*)d_in[0];
    const int*   seqlen   = (const int*)  d_in[1];
    const float* W1w      = (const float*)d_in[2];
    const float* W1b      = (const float*)d_in[3];
    const float* W2w      = (const float*)d_in[4];
    const float* W2b      = (const float*)d_in[5];
    float*       out      = (float*)d_out;

    const int N = in_sizes[1];   // one wave32 workgroup per sample
    mie_kernel<<<N, 32, 0, stream>>>(behavior, seqlen, W1w, W1b, W2w, W2b, out);
}
